// EPCPoseModel_67104569032876
// MI455X (gfx1250) — matrix operand order repaired and checked by reading, not compiled, not run on hardware
//
#include <hip/hip_runtime.h>
#include <hip/hip_bf16.h>

typedef __attribute__((ext_vector_type(16))) _Float16 v16h;
typedef __attribute__((ext_vector_type(8)))  float    v8f;

#define NJOINT 13
#define MTOT   128
#define ROWS   (MTOT * NJOINT)   /* 1664 */
#define NPTS   7500
#define NCH    28
#define NS     500
#define NSP    512
#define BN_EPS 1e-5f

/* d_out layout (floats): inp[8192] | aw[832000] | hs[1536] | cs[1536] */
#define OUT_AW_OFF 8192
#define OUT_HS_OFF 840192
#define OUT_CS_OFF 841728

/* d_ws layout in 4-byte elements */
#define WSE_VEC  ((size_t)ROWS * NS)                 /* after idx ints   */
#define WSE_FEAT (WSE_VEC + (size_t)ROWS * 64)
#define WSE_SEQA (WSE_FEAT + (size_t)MTOT * 64)
#define WSE_SEQB (WSE_SEQA + (size_t)MTOT * 64)

/* ----------------------------------------------------------------------- */
/* WMMA helpers: CDNA5 wave32 16x16x32 f16 -> f32                           */
/* ----------------------------------------------------------------------- */
__device__ __forceinline__ v8f wmma16x16x32(v16h a, v16h b, v8f c) {
#if defined(__gfx1250__)
  /* (neg_a, A, neg_b, B, c_mod, C, reuse_a, reuse_b) */
  return __builtin_amdgcn_wmma_f32_16x16x32_f16(false, a, false, b, (short)0, c,
                                                false, false);
#else
  (void)a; (void)b;
  return c;
#endif
}

/* Per-lane-packed operand tiles: each lane's 16 halves are contiguous, so an
   operand load is one 32-byte LDS read per lane (2x ds_load_b128).

   A-tile (16x32, MxK), lane L: row = L&15,
     L<16 : halves 0..7 -> K 0..7,  8..15 -> K 16..23
     L>=16: halves 0..7 -> K 8..15, 8..15 -> K 24..31
   swizA(r,k) = element offset inside a 512-half tile.                       */
__device__ __forceinline__ int swizA(int r, int k) {
  return ((r + ((k & 8) ? 16 : 0)) << 4) + ((k & 7) + ((k & 16) ? 8 : 0));
}
/* B-tile (32x16, KxN), lane L: col = L&15, same K mapping as the A-tile */
__device__ __forceinline__ int swizB(int k, int n15) {
  return ((n15 + ((k & 8) ? 16 : 0)) << 4) + ((k & 7) + ((k & 16) ? 8 : 0));
}
__device__ __forceinline__ v16h lds_tile16(const _Float16* base, int lane) {
  return *(const v16h*)(base + ((size_t)lane << 4));
}

/* ----------------------------------------------------------------------- */
/* Kernel 1: per-(frame,joint) exact sorted top-500 by squared 2D distance  */
/* ----------------------------------------------------------------------- */
__global__ __launch_bounds__(256) void topk_kernel(const float* __restrict__ x,
                                                   const float* __restrict__ kpt,
                                                   int* __restrict__ idx_out) {
  __shared__ unsigned long long keys[NPTS];
  __shared__ unsigned long long red[256];
  const int row = blockIdx.x;
  const int tid = threadIdx.x;
  const int m   = row / NJOINT;
  const float ax = kpt[row * 2 + 0];
  const float ay = kpt[row * 2 + 1];
  const float* xm = x + (size_t)m * NPTS * NCH;

  for (int i = tid; i < NPTS; i += 256) {
    if (i + 256 < NPTS) __builtin_prefetch(&xm[(size_t)(i + 256) * NCH], 0, 1);
    const float px = xm[(size_t)i * NCH + 0];
    const float py = xm[(size_t)i * NCH + 1];
    /* match reference formula: |a|^2 + |p|^2 - 2 a.p */
    const float d = ax * ax + ay * ay + px * px + py * py
                    - 2.0f * (ax * px + ay * py);
    unsigned u = __float_as_uint(d);
    u ^= (u >> 31) ? 0xFFFFFFFFu : 0x80000000u; /* order-preserving map */
    keys[i] = ((unsigned long long)u << 32) | (unsigned)i;
  }
  __syncthreads();

  for (int k = 0; k < NS; ++k) {
    unsigned long long lm = 0xFFFFFFFFFFFFFFFFull;
    for (int i = tid; i < NPTS; i += 256) {
      const unsigned long long v = keys[i];
      lm = (v < lm) ? v : lm;
    }
    red[tid] = lm;
    __syncthreads();
    for (int s = 128; s > 0; s >>= 1) {
      if (tid < s) {
        const unsigned long long o = red[tid + s];
        if (o < red[tid]) red[tid] = o;
      }
      __syncthreads();
    }
    if (tid == 0) {
      const unsigned long long w = red[0];
      const int arg = (int)(unsigned)(w & 0xFFFFFFFFull);
      idx_out[(size_t)row * NS + k] = arg;
      keys[arg] = 0xFFFFFFFFFFFFFFFFull;
    }
    __syncthreads();
  }
}

/* ----------------------------------------------------------------------- */
/* Kernel 2: point MLP (WMMA) + attention softmax + weighted sum            */
/* ----------------------------------------------------------------------- */
struct ApnArgs {
  const float* x;
  const float* kpt;
  const int*   idx;
  const float *w1, *b1, *g1, *be1, *mu1, *va1;
  const float *w2, *b2, *g2, *be2, *mu2, *va2;
  const float *w3, *b3, *g3, *be3, *mu3, *va3;
  const float *wa, *ba;
  float* aw_out;   /* [ROWS*NS] */
  float* vec_out;  /* [ROWS*64] */
};

/* Build the 16x32 feature A-operand directly in registers: each lane owns one
   row and a fixed K-subset per the wave32 A layout (no LDS round-trip). */
__device__ __forceinline__ v16h build_featA(int t, int lane,
                                            const float* __restrict__ xm,
                                            const int* __restrict__ idxRow,
                                            float ax, float ay) {
  v16h a;
#pragma unroll
  for (int h = 0; h < 16; ++h) a[h] = (_Float16)0.0f;
  const int r  = lane & 15;
  const int gr = t * 16 + r;
  if (gr < NS) {
    const float* p = xm + (size_t)idxRow[gr] * NCH;
    const int kbase = (lane < 16) ? 0 : 8;
#pragma unroll
    for (int h = 0; h < 16; ++h) {
      const int k = kbase + (h & 7) + ((h & 8) ? 16 : 0);
      float val;
      if      (k == 0) val = ax;
      else if (k == 1) val = ay;
      else if (k == 2) val = p[0] - ax;
      else if (k == 3) val = p[1] - ay;
      else if (k <= 28) val = p[k - 1];   /* gpts: channels 3..27 */
      else              val = p[k - 29];  /* gxyz: channels 0..2  */
      a[h] = (_Float16)val;
    }
  }
  return a;
}

/* One 16-row tile through the 3 MLP layers; h3 (16x64 f16, row-major) in sB.
   All staging is per-wave; wave-order LDS semantics make barriers unneeded. */
__device__ __forceinline__ void apn_tile(
    int t, int lane, const float* __restrict__ xm, const int* __restrict__ idxRow,
    float ax, float ay, const _Float16* w1s, const _Float16* w2s,
    const _Float16* w3s, const float* fb1, const float* fb2, const float* fb3,
    _Float16* sA, _Float16* sB) {
  const int n15 = lane & 15;
  const int mb  = (lane < 16) ? 0 : 8;

  /* ---- layer 1: 32 -> 32 (A in registers, 2 N-tiles) ---- */
  {
    const v16h A = build_featA(t, lane, xm, idxRow, ax, ay);
#pragma unroll
    for (int nt = 0; nt < 2; ++nt) {
      v8f acc = {};
      const v16h B = lds_tile16(w1s + nt * 512, lane);
      acc = wmma16x16x32(A, B, acc);
      const int n = n15 + nt * 16;
#pragma unroll
      for (int r = 0; r < 8; ++r) {
        const float vv = acc[r] + fb1[n];
        sB[swizA(r + mb, n)] = (_Float16)fmaxf(vv, 0.0f);  /* h1, A-swizzled */
      }
    }
  }
  /* ---- layer 2: 32 -> 48 (A = h1 from sB subtile 0, 3 N-tiles) ---- */
  {
    const v16h A = lds_tile16(sB, lane);
#pragma unroll
    for (int nt = 0; nt < 3; ++nt) {
      v8f acc = {};
      const v16h B = lds_tile16(w2s + nt * 512, lane);
      acc = wmma16x16x32(A, B, acc);
      const int n = n15 + nt * 16;
#pragma unroll
      for (int r = 0; r < 8; ++r) {
        const float vv = acc[r] + fb2[n];
        /* h2, A-swizzled across 2 K-subtiles; subtile1 k'>=16 stays zero */
        sA[(n >> 5) * 512 + swizA(r + mb, n & 31)] = (_Float16)fmaxf(vv, 0.0f);
      }
    }
  }
  /* ---- layer 3: 48 -> 64 (K padded to 64; w3s rows 48..63 zero) ---- */
  {
    const v16h A0 = lds_tile16(sA, lane);
    const v16h A1 = lds_tile16(sA + 512, lane);
#pragma unroll
    for (int nt = 0; nt < 4; ++nt) {
      v8f acc = {};
      acc = wmma16x16x32(A0, lds_tile16(w3s + (0 * 4 + nt) * 512, lane), acc);
      acc = wmma16x16x32(A1, lds_tile16(w3s + (1 * 4 + nt) * 512, lane), acc);
      const int n = n15 + nt * 16;
#pragma unroll
      for (int r = 0; r < 8; ++r) {
        const float vv = acc[r] + fb3[n];
        sB[(r + mb) * 64 + n] = (_Float16)fmaxf(vv, 0.0f);  /* h3, row-major */
      }
    }
  }
}

__global__ __launch_bounds__(256) void apn_attn_kernel(ApnArgs P) {
  __shared__ alignas(32) _Float16 w1s[2 * 512];   /* 2 B-tiles   */
  __shared__ alignas(32) _Float16 w2s[3 * 512];   /* 3 B-tiles   */
  __shared__ alignas(32) _Float16 w3s[8 * 512];   /* 2 K x 4 N   */
  __shared__ alignas(32) _Float16 stage[8][2][1024];
  __shared__ float fb1[32], fb2[48], fb3[64], was[64];
  __shared__ float logits[NSP];
  __shared__ float vecs[64];
  __shared__ float redf[256];

  const int row  = blockIdx.x;
  const int tid  = threadIdx.x;
  const int lane = tid & 31;
  const int wv   = tid >> 5;
  const int m    = row / NJOINT;
  const float ax = P.kpt[row * 2 + 0];
  const float ay = P.kpt[row * 2 + 1];
  const float* xm     = P.x + (size_t)m * NPTS * NCH;
  const int*   idxRow = P.idx + (size_t)row * NS;
  const float  bav    = P.ba[0];

  /* ---- fold BN into f16 weights (stored pre-swizzled) + fused biases ---- */
  for (int e = tid; e < 32; e += 256) {
    const float s = P.g1[e] * rsqrtf(P.va1[e] + BN_EPS);
    fb1[e] = (P.b1[e] - P.mu1[e]) * s + P.be1[e];
  }
  for (int e = tid; e < 48; e += 256) {
    const float s = P.g2[e] * rsqrtf(P.va2[e] + BN_EPS);
    fb2[e] = (P.b2[e] - P.mu2[e]) * s + P.be2[e];
  }
  for (int e = tid; e < 64; e += 256) {
    const float s = P.g3[e] * rsqrtf(P.va3[e] + BN_EPS);
    fb3[e] = (P.b3[e] - P.mu3[e]) * s + P.be3[e];
    was[e] = P.wa[e];
  }
  for (int e = tid; e < 32 * 32; e += 256) {
    const int k = e >> 5, n = e & 31;
    const float s = P.g1[n] * rsqrtf(P.va1[n] + BN_EPS);
    w1s[(n >> 4) * 512 + swizB(k, n & 15)] = (_Float16)(P.w1[e] * s);
  }
  for (int e = tid; e < 32 * 48; e += 256) {
    const int k = e / 48, n = e - k * 48;
    const float s = P.g2[n] * rsqrtf(P.va2[n] + BN_EPS);
    w2s[(n >> 4) * 512 + swizB(k, n & 15)] = (_Float16)(P.w2[e] * s);
  }
  for (int e = tid; e < 64 * 64; e += 256) {
    const int k = e >> 6, n = e & 63;
    float v = 0.0f;
    if (k < 48) {
      const float s = P.g3[n] * rsqrtf(P.va3[n] + BN_EPS);
      v = P.w3[k * 64 + n] * s;
    }
    w3s[((k >> 5) * 4 + (n >> 4)) * 512 + swizB(k & 31, n & 15)] = (_Float16)v;
  }

  _Float16* sA = &stage[wv][0][0];
  _Float16* sB = &stage[wv][1][0];
  /* zero per-wave staging once (keeps never-written K-pad slots finite) */
  for (int e = lane; e < 1024; e += 32) { sA[e] = (_Float16)0.0f; sB[e] = (_Float16)0.0f; }
  __syncthreads();

  /* ---- pass 1: logits for all 500 samples (32 tiles, 4 per wave) ---- */
  for (int i = 0; i < 4; ++i) {
    const int t = wv * 4 + i;
    apn_tile(t, lane, xm, idxRow, ax, ay, w1s, w2s, w3s, fb1, fb2, fb3, sA, sB);
    if (lane < 16) {
      float acc = bav;
      const _Float16* hrow = sB + lane * 64;
#pragma unroll 8
      for (int c = 0; c < 64; ++c) acc += (float)hrow[c] * was[c];
      logits[t * 16 + lane] = acc;
    }
  }
  __syncthreads();

  /* ---- softmax over samples 0..499 ---- */
  float lm = -3.402823e38f;
  for (int i = tid; i < NS; i += 256) lm = fmaxf(lm, logits[i]);
  redf[tid] = lm;
  __syncthreads();
  for (int s = 128; s > 0; s >>= 1) {
    if (tid < s) redf[tid] = fmaxf(redf[tid], redf[tid + s]);
    __syncthreads();
  }
  const float mx = redf[0];
  __syncthreads();
  float ls = 0.0f;
  for (int i = tid; i < NS; i += 256) ls += expf(logits[i] - mx);
  redf[tid] = ls;
  __syncthreads();
  for (int s = 128; s > 0; s >>= 1) {
    if (tid < s) redf[tid] += redf[tid + s];
    __syncthreads();
  }
  const float inv = 1.0f / redf[0];
  __syncthreads();
  for (int i = tid; i < NSP; i += 256) {
    float a = 0.0f;
    if (i < NS) {
      a = expf(logits[i] - mx) * inv;
      P.aw_out[(size_t)row * NS + i] = a;
    }
    logits[i] = a; /* reuse buffer as attention weights (pad rows = 0) */
  }
  if (tid < 64) vecs[tid] = 0.0f;
  __syncthreads();

  /* ---- pass 2: recompute h, accumulate vec = sum_i aw_i * h_i ---- */
  for (int i = 0; i < 4; ++i) {
    const int t = wv * 4 + i;
    apn_tile(t, lane, xm, idxRow, ax, ay, w1s, w2s, w3s, fb1, fb2, fb3, sA, sB);
#pragma unroll
    for (int cc = lane; cc < 64; cc += 32) {
      float part = 0.0f;
#pragma unroll 4
      for (int r = 0; r < 16; ++r)
        part += logits[t * 16 + r] * (float)sB[r * 64 + cc];
      atomicAdd(&vecs[cc], part);
    }
  }
  __syncthreads();
  if (tid < 64) P.vec_out[(size_t)row * 64 + tid] = vecs[tid];
}

/* ----------------------------------------------------------------------- */
/* Kernel 3: 1D conv chain 64->96->128->64 over joint axis (13->9->5->1)    */
/* ----------------------------------------------------------------------- */
struct ConvArgs {
  const float* vec; /* [M*13, 64] */
  const float *w1, *b1, *g1, *be1, *mu1, *va1;
  const float *w2, *b2, *g2, *be2, *mu2, *va2;
  const float *w3, *b3, *g3, *be3, *mu3, *va3;
  float* feat; /* [M, 64] */
};

__global__ __launch_bounds__(256) void aconv_kernel(ConvArgs P) {
  __shared__ float fin[64 * 13];
  __shared__ float o1[96 * 9];
  __shared__ float o2[128 * 5];
  const int m = blockIdx.x, tid = threadIdx.x;

  for (int e = tid; e < 64 * 13; e += 256) {
    const int c = e / 13, j = e - c * 13;
    fin[e] = P.vec[((size_t)m * NJOINT + j) * 64 + c];
  }
  __syncthreads();

  for (int e = tid; e < 96 * 9; e += 256) {
    const int o = e / 9, t = e - o * 9;
    float acc = 0.0f;
    const float* w = P.w1 + (size_t)o * 64 * 5;
    for (int c = 0; c < 64; ++c)
#pragma unroll
      for (int k = 0; k < 5; ++k) acc += w[c * 5 + k] * fin[c * 13 + t + k];
    const float s = P.g1[o] * rsqrtf(P.va1[o] + BN_EPS);
    const float y = (acc + P.b1[o] - P.mu1[o]) * s + P.be1[o];
    o1[e] = fmaxf(y, 0.0f);
  }
  __syncthreads();

  for (int e = tid; e < 128 * 5; e += 256) {
    const int o = e / 5, t = e - o * 5;
    float acc = 0.0f;
    const float* w = P.w2 + (size_t)o * 96 * 5;
    for (int c = 0; c < 96; ++c)
#pragma unroll
      for (int k = 0; k < 5; ++k) acc += w[c * 5 + k] * o1[c * 9 + t + k];
    const float s = P.g2[o] * rsqrtf(P.va2[o] + BN_EPS);
    const float y = (acc + P.b2[o] - P.mu2[o]) * s + P.be2[o];
    o2[e] = fmaxf(y, 0.0f);
  }
  __syncthreads();

  for (int o = tid; o < 64; o += 256) {
    float acc = 0.0f;
    const float* w = P.w3 + (size_t)o * 128 * 5;
    for (int c = 0; c < 128; ++c)
#pragma unroll
      for (int k = 0; k < 5; ++k) acc += w[c * 5 + k] * o2[c * 5 + k];
    const float s = P.g3[o] * rsqrtf(P.va3[o] + BN_EPS);
    const float y = (acc + P.b3[o] - P.mu3[o]) * s + P.be3[o];
    P.feat[(size_t)m * 64 + o] = fmaxf(y, 0.0f);
  }
}

/* ----------------------------------------------------------------------- */
/* Kernel 4: 3-layer LSTM, B=8, T=16, H=64 (single block, sequential)       */
/* ----------------------------------------------------------------------- */
struct LstmArgs {
  const float* feat;
  const float* h0;
  const float* c0;
  const float *Wih0, *Whh0, *bih0, *bhh0;
  const float *Wih1, *Whh1, *bih1, *bhh1;
  const float *Wih2, *Whh2, *bih2, *bhh2;
  float* seqA;
  float* seqB;
  float* out_inp;
  float* out_hs;
  float* out_cs;
};

__device__ __forceinline__ float sigmf(float v) { return 1.0f / (1.0f + expf(-v)); }

__global__ __launch_bounds__(256) void lstm_kernel(LstmArgs P) {
  __shared__ float hsS[8][64];
  __shared__ float csS[8][64];
  __shared__ float xb[8][64];
  __shared__ float zb[8][256];
  const int tid = threadIdx.x;
  const float* WihA[3] = {P.Wih0, P.Wih1, P.Wih2};
  const float* WhhA[3] = {P.Whh0, P.Whh1, P.Whh2};
  const float* bihA[3] = {P.bih0, P.bih1, P.bih2};
  const float* bhhA[3] = {P.bhh0, P.bhh1, P.bhh2};

  for (int l = 0; l < 3; ++l) {
    const float* Wih = WihA[l];
    const float* Whh = WhhA[l];
    const float* bih = bihA[l];
    const float* bhh = bhhA[l];
    const float* in  = (l == 0) ? P.feat : ((l == 1) ? P.seqA : P.seqB);
    float* outp      = (l == 0) ? P.seqA : ((l == 1) ? P.seqB : P.out_inp);

    for (int e = tid; e < 512; e += 256) {
      hsS[e >> 6][e & 63] = P.h0[(size_t)l * 512 + e];
      csS[e >> 6][e & 63] = P.c0[(size_t)l * 512 + e];
    }
    __syncthreads();

    for (int t = 0; t < 16; ++t) {
      for (int e = tid; e < 512; e += 256)
        xb[e >> 6][e & 63] = in[((size_t)(e >> 6) * 16 + t) * 64 + (e & 63)];
      __syncthreads();
      for (int e = tid; e < 2048; e += 256) {
        const int b = e >> 8, u = e & 255;
        float acc = bih[u] + bhh[u];
        const float* wi = Wih + (size_t)u * 64;
        const float* wh = Whh + (size_t)u * 64;
        for (int k = 0; k < 64; ++k) acc += xb[b][k] * wi[k] + hsS[b][k] * wh[k];
        zb[b][u] = acc;
      }
      __syncthreads();
      for (int e = tid; e < 512; e += 256) {
        const int b = e >> 6, u = e & 63;
        const float zi = zb[b][u], zf = zb[b][u + 64];
        const float zg = zb[b][u + 128], zo = zb[b][u + 192];
        const float cn = sigmf(zf) * csS[b][u] + sigmf(zi) * tanhf(zg);
        const float hn = sigmf(zo) * tanhf(cn);
        csS[b][u] = cn;
        hsS[b][u] = hn;
        outp[((size_t)b * 16 + t) * 64 + u] = hn;
      }
      __syncthreads();
    }
    for (int e = tid; e < 512; e += 256) {
      P.out_hs[(size_t)l * 512 + e] = hsS[e >> 6][e & 63];
      P.out_cs[(size_t)l * 512 + e] = csS[e >> 6][e & 63];
    }
    __syncthreads();
  }
}

/* ----------------------------------------------------------------------- */
extern "C" void kernel_launch(void* const* d_in, const int* in_sizes, int n_in,
                              void* d_out, int out_size, void* d_ws, size_t ws_size,
                              hipStream_t stream) {
  (void)in_sizes; (void)n_in; (void)out_size; (void)ws_size;

  const float* x   = (const float*)d_in[0];
  const float* kpt = (const float*)d_in[1];
  const float* h0  = (const float*)d_in[2];
  const float* c0  = (const float*)d_in[3];
  /* d_in[4..6]: B, L, feature_size (compile-time constants here) */

  /* apn params */
  const float* a_w1  = (const float*)d_in[7];
  const float* a_b1  = (const float*)d_in[8];
  const float* a_g1  = (const float*)d_in[9];
  const float* a_be1 = (const float*)d_in[10];
  const float* a_mu1 = (const float*)d_in[11];
  const float* a_va1 = (const float*)d_in[12];
  const float* a_w2  = (const float*)d_in[13];
  const float* a_b2  = (const float*)d_in[14];
  const float* a_g2  = (const float*)d_in[15];
  const float* a_be2 = (const float*)d_in[16];
  const float* a_mu2 = (const float*)d_in[17];
  const float* a_va2 = (const float*)d_in[18];
  const float* a_w3  = (const float*)d_in[19];
  const float* a_b3  = (const float*)d_in[20];
  const float* a_g3  = (const float*)d_in[21];
  const float* a_be3 = (const float*)d_in[22];
  const float* a_mu3 = (const float*)d_in[23];
  const float* a_va3 = (const float*)d_in[24];
  const float* a_wa  = (const float*)d_in[25];
  const float* a_ba  = (const float*)d_in[26];
  /* aconv params */
  const float* c_w1  = (const float*)d_in[27];
  const float* c_b1  = (const float*)d_in[28];
  const float* c_g1  = (const float*)d_in[29];
  const float* c_be1 = (const float*)d_in[30];
  const float* c_mu1 = (const float*)d_in[31];
  const float* c_va1 = (const float*)d_in[32];
  const float* c_w2  = (const float*)d_in[33];
  const float* c_b2  = (const float*)d_in[34];
  const float* c_g2  = (const float*)d_in[35];
  const float* c_be2 = (const float*)d_in[36];
  const float* c_mu2 = (const float*)d_in[37];
  const float* c_va2 = (const float*)d_in[38];
  const float* c_w3  = (const float*)d_in[39];
  const float* c_b3  = (const float*)d_in[40];
  const float* c_g3  = (const float*)d_in[41];
  const float* c_be3 = (const float*)d_in[42];
  const float* c_mu3 = (const float*)d_in[43];
  const float* c_va3 = (const float*)d_in[44];
  /* lstm params */
  const float* l_Wih0 = (const float*)d_in[45];
  const float* l_Whh0 = (const float*)d_in[46];
  const float* l_bih0 = (const float*)d_in[47];
  const float* l_bhh0 = (const float*)d_in[48];
  const float* l_Wih1 = (const float*)d_in[49];
  const float* l_Whh1 = (const float*)d_in[50];
  const float* l_bih1 = (const float*)d_in[51];
  const float* l_bhh1 = (const float*)d_in[52];
  const float* l_Wih2 = (const float*)d_in[53];
  const float* l_Whh2 = (const float*)d_in[54];
  const float* l_bih2 = (const float*)d_in[55];
  const float* l_bhh2 = (const float*)d_in[56];

  int*   idx  = (int*)d_ws;
  float* base = (float*)d_ws;
  float* vec  = base + WSE_VEC;
  float* feat = base + WSE_FEAT;
  float* seqA = base + WSE_SEQA;
  float* seqB = base + WSE_SEQB;
  float* out  = (float*)d_out;

  topk_kernel<<<ROWS, 256, 0, stream>>>(x, kpt, idx);

  ApnArgs ap;
  ap.x = x; ap.kpt = kpt; ap.idx = idx;
  ap.w1 = a_w1; ap.b1 = a_b1; ap.g1 = a_g1; ap.be1 = a_be1; ap.mu1 = a_mu1; ap.va1 = a_va1;
  ap.w2 = a_w2; ap.b2 = a_b2; ap.g2 = a_g2; ap.be2 = a_be2; ap.mu2 = a_mu2; ap.va2 = a_va2;
  ap.w3 = a_w3; ap.b3 = a_b3; ap.g3 = a_g3; ap.be3 = a_be3; ap.mu3 = a_mu3; ap.va3 = a_va3;
  ap.wa = a_wa; ap.ba = a_ba;
  ap.aw_out = out + OUT_AW_OFF; ap.vec_out = vec;
  apn_attn_kernel<<<ROWS, 256, 0, stream>>>(ap);

  ConvArgs cp;
  cp.vec = vec;
  cp.w1 = c_w1; cp.b1 = c_b1; cp.g1 = c_g1; cp.be1 = c_be1; cp.mu1 = c_mu1; cp.va1 = c_va1;
  cp.w2 = c_w2; cp.b2 = c_b2; cp.g2 = c_g2; cp.be2 = c_be2; cp.mu2 = c_mu2; cp.va2 = c_va2;
  cp.w3 = c_w3; cp.b3 = c_b3; cp.g3 = c_g3; cp.be3 = c_be3; cp.mu3 = c_mu3; cp.va3 = c_va3;
  cp.feat = feat;
  aconv_kernel<<<MTOT, 256, 0, stream>>>(cp);

  LstmArgs lp;
  lp.feat = feat; lp.h0 = h0; lp.c0 = c0;
  lp.Wih0 = l_Wih0; lp.Whh0 = l_Whh0; lp.bih0 = l_bih0; lp.bhh0 = l_bhh0;
  lp.Wih1 = l_Wih1; lp.Whh1 = l_Whh1; lp.bih1 = l_bih1; lp.bhh1 = l_bhh1;
  lp.Wih2 = l_Wih2; lp.Whh2 = l_Whh2; lp.bih2 = l_bih2; lp.bhh2 = l_bhh2;
  lp.seqA = seqA; lp.seqB = seqB;
  lp.out_inp = out; lp.out_hs = out + OUT_HS_OFF; lp.out_cs = out + OUT_CS_OFF;
  lstm_kernel<<<1, 256, 0, stream>>>(lp);
}
